// Router_19095424598754
// MI455X (gfx1250) — matrix operand order repaired and checked by reading, not compiled, not run on hardware
//
#include <hip/hip_runtime.h>

// ---- problem constants (match reference setup_inputs) ----
#define G_ 8
#define T_ 4096
#define D_ 2048
#define E_ 64
#define NTOK (G_ * T_)                    // 32768 tokens
static constexpr size_t GTE = (size_t)NTOK * E_;   // 2097152
#define LDS_ROW 2056                      // bf16 elems per padded weight row
#define ROWB (LDS_ROW * 2)                // 4112 bytes
#define SMEM_BYTES (E_ * ROWB)            // 263168 B  (<320KB WGP LDS)

typedef __attribute__((ext_vector_type(16))) __bf16 v16bf;
typedef __attribute__((ext_vector_type(8)))  float  v8f;
typedef __attribute__((ext_vector_type(4)))  float  f4;   // clang ext vector (nontemporal-compatible)

union AFrag { v16bf v; __bf16 h[16]; };
union BFrag { v16bf v; uint4 q[2]; };

__global__ void zero_z(float* p) { if (threadIdx.x == 0) *p = 0.f; }

__global__ void __launch_bounds__(256)
router_fused(const float* __restrict__ tok,
             const float* __restrict__ noi,
             const float* __restrict__ wgt,
             const float* __restrict__ bias,
             float* __restrict__ out)
{
    extern __shared__ __align__(16) unsigned char smem[];
    const int tid  = threadIdx.x;
    const int lane = tid & 31;
    const int wid  = tid >> 5;

    // ---- stage weight: fp32 global -> bf16 LDS (regular-temporal: keep L2-hot) ----
    // Interleaved rows: expert n -> physical row ((n&15)*4 + (n>>4)), so the 4
    // expert tiles a lane reads sit at +0/+4112/+8224/+12336 bytes (DS offset imm).
    {
        const f4* w4 = (const f4*)wgt;
        for (int c = tid; c < (E_ * D_) / 4; c += 256) {
            int n = c >> 9;                 // 512 float4 per 2048-wide row
            int k = (c & 511) << 2;
            int prow = ((n & 15) << 2) | (n >> 4);
            f4 w = w4[c];
            __bf16 h[4] = { (__bf16)w.x, (__bf16)w.y, (__bf16)w.z, (__bf16)w.w };
            *(uint2*)(smem + (size_t)prow * ROWB + (size_t)k * 2) = *(const uint2*)h;
        }
    }
    __syncthreads();

    // ---- per-lane WMMA fragment coordinates (ISA 7.12.2 layouts, wave32) ----
    const int m  = lane & 15;   // A row / B column-low / C column-low
    const int kh = lane >> 4;   // K-half selector
    const int tokBase = (blockIdx.x * 8 + wid) * 16;
    const size_t row = (size_t)(tokBase + m) * D_;
    const f4* tp = (const f4*)(tok + row);
    const f4* np = (const f4*)(noi + row);

    v8f acc[4] = { {0.f}, {0.f}, {0.f}, {0.f} };
    float ab0 = 0.f, ab1 = 0.f, ab2 = 0.f, ab3 = 0.f;   // split |x| chains

    // lane's B base: physical row 4*m, K-half kh; bumps +64B per k-step
    const unsigned char* bp = smem + (unsigned)(4 * m) * ROWB + (unsigned)kh * 16;

    // ---- fused jitter + GEMM main loop: 64 k-steps of 32, 4 WMMAs each ----
    // token/noise are consumed exactly once -> non-temporal streaming loads.
    #pragma unroll 4
    for (int i = 0; i < D_ / 32; ++i) {
        const int c0 = 8 * i + 2 * kh;
        f4 t0 = __builtin_nontemporal_load(tp + c0);
        f4 t1 = __builtin_nontemporal_load(tp + c0 + 1);
        f4 t2 = __builtin_nontemporal_load(tp + c0 + 4);
        f4 t3 = __builtin_nontemporal_load(tp + c0 + 5);
        f4 u0 = __builtin_nontemporal_load(np + c0);
        f4 u1 = __builtin_nontemporal_load(np + c0 + 1);
        f4 u2 = __builtin_nontemporal_load(np + c0 + 4);
        f4 u3 = __builtin_nontemporal_load(np + c0 + 5);

        AFrag a;
        #define JIT(idx, tv, uv, abx) { float xv = (tv) * (0.99f + 0.02f * (uv)); \
                                        abx += fabsf(xv); a.h[idx] = (__bf16)xv; }
        JIT(0,  t0.x, u0.x, ab0) JIT(1,  t0.y, u0.y, ab1)
        JIT(2,  t0.z, u0.z, ab2) JIT(3,  t0.w, u0.w, ab3)
        JIT(4,  t1.x, u1.x, ab0) JIT(5,  t1.y, u1.y, ab1)
        JIT(6,  t1.z, u1.z, ab2) JIT(7,  t1.w, u1.w, ab3)
        JIT(8,  t2.x, u2.x, ab0) JIT(9,  t2.y, u2.y, ab1)
        JIT(10, t2.z, u2.z, ab2) JIT(11, t2.w, u2.w, ab3)
        JIT(12, t3.x, u3.x, ab0) JIT(13, t3.y, u3.y, ab1)
        JIT(14, t3.z, u3.z, ab2) JIT(15, t3.w, u3.w, ab3)
        #undef JIT

        #pragma unroll
        for (int t = 0; t < 4; ++t) {
            BFrag b;
            b.q[0] = *(const uint4*)(bp + t * ROWB);
            b.q[1] = *(const uint4*)(bp + t * ROWB + 32);
            acc[t] = __builtin_amdgcn_wmma_f32_16x16x32_bf16(
                         false, a.v, false, b.v, (short)0, acc[t], false, false);
        }
        bp += 64;   // 32 bf16 per k-step
    }

    // ---- epilogue: softmax / mask / logsumexp / z-loss for 16 tokens x 64 experts ----
    float absacc  = (ab0 + ab1) + (ab2 + ab3);
    float absfull = absacc + __shfl_xor(absacc, 16);   // full-row |x| sum for token m
    float b0 = bias[m], b1 = bias[m + 16], b2 = bias[m + 32], b3 = bias[m + 48];
    float zacc = 0.f;

    #pragma unroll
    for (int j = 0; j < 8; ++j) {
        float v0 = acc[0][j] + b0;
        float v1 = acc[1][j] + b1;
        float v2 = acc[2][j] + b2;
        float v3 = acc[3][j] + b3;
        const int tokM = j + 8 * kh;                   // C layout: VGPR j -> row j / j+8

        // softmax over 64 experts (4 local values x 16 lanes per half-wave)
        float mx = fmaxf(fmaxf(v0, v1), fmaxf(v2, v3));
        #pragma unroll
        for (int d = 1; d <= 8; d <<= 1) mx = fmaxf(mx, __shfl_xor(mx, d));
        float e0 = __expf(v0 - mx), e1 = __expf(v1 - mx);
        float e2 = __expf(v2 - mx), e3 = __expf(v3 - mx);
        float s = e0 + e1 + e2 + e3;
        #pragma unroll
        for (int d = 1; d <= 8; d <<= 1) s += __shfl_xor(s, d);
        float inv = 1.f / s;

        // outputs are write-once, never re-read -> non-temporal stores
        const size_t obase = (size_t)(tokBase + tokM) * E_ + m;
        __builtin_nontemporal_store(e0 * inv, out + obase);
        __builtin_nontemporal_store(e1 * inv, out + obase + 16);
        __builtin_nontemporal_store(e2 * inv, out + obase + 32);
        __builtin_nontemporal_store(e3 * inv, out + obase + 48);

        // padding mask + masked logits
        float ab = __shfl(absfull, tokM);              // lane tokM holds token tokM's sum
        float mk = (ab > 0.f) ? 1.f : 0.f;
        float w0 = v0 * mk, w1 = v1 * mk, w2 = v2 * mk, w3 = v3 * mk;
        float* mo = out + GTE;
        __builtin_nontemporal_store(w0, mo + obase);
        __builtin_nontemporal_store(w1, mo + obase + 16);
        __builtin_nontemporal_store(w2, mo + obase + 32);
        __builtin_nontemporal_store(w3, mo + obase + 48);

        // logsumexp over masked logits
        float mx2 = fmaxf(fmaxf(w0, w1), fmaxf(w2, w3));
        #pragma unroll
        for (int d = 1; d <= 8; d <<= 1) mx2 = fmaxf(mx2, __shfl_xor(mx2, d));
        float s2 = __expf(w0 - mx2) + __expf(w1 - mx2) + __expf(w2 - mx2) + __expf(w3 - mx2);
        #pragma unroll
        for (int d = 1; d <= 8; d <<= 1) s2 += __shfl_xor(s2, d);
        float lz = mx2 + __logf(s2);
        zacc += ((lane & 15) == 0) ? lz * lz : 0.f;    // one contribution per token row
    }

    zacc += __shfl_xor(zacc, 16);
    if (lane == 0)
        unsafeAtomicAdd(out + 2 * GTE, zacc * (1.f / (float)NTOK));
}

extern "C" void kernel_launch(void* const* d_in, const int* in_sizes, int n_in,
                              void* d_out, int out_size, void* d_ws, size_t ws_size,
                              hipStream_t stream) {
    (void)in_sizes; (void)n_in; (void)out_size; (void)d_ws; (void)ws_size;
    const float* tok  = (const float*)d_in[0];
    const float* noi  = (const float*)d_in[1];
    const float* wgt  = (const float*)d_in[2];
    const float* bias = (const float*)d_in[3];
    float* out = (float*)d_out;

    static_assert(SMEM_BYTES <= 320 * 1024, "LDS over WGP limit");
    (void)hipFuncSetAttribute((const void*)router_fused,
                              hipFuncAttributeMaxDynamicSharedMemorySize, SMEM_BYTES);

    zero_z<<<1, 32, 0, stream>>>(out + 2 * GTE);
    router_fused<<<NTOK / 128, 256, SMEM_BYTES, stream>>>(tok, noi, wgt, bias, out);
}
